// LongformerSelfAttention_11733850653008
// MI455X (gfx1250) — matrix operand order repaired and checked by reading, not compiled, not run on hardware
//
#include <hip/hip_runtime.h>
#include <hip/hip_bf16.h>
#include <cstdint>

typedef __attribute__((ext_vector_type(16))) _Float16 v16h;
typedef __attribute__((ext_vector_type(8)))  _Float16 v8h;
typedef __attribute__((ext_vector_type(8)))  float    v8f;

#define H_  12
#define S_  4096
#define D_  768
#define DH  64
#define W_  128
#define G_  32
#define NC  (S_ / W_)   // 32 chunks

// ---------------------------------------------------------------------------
// Fragment helpers for V_WMMA_F32_16X16X32_F16 (wave32).
// A layout: lane row = lane&15, K-base = (lane>>4)*16, halves {kb2..kb2+15}
// B layout: lane col = lane&15, K-base = (lane>>4)*16, halves {kb2..kb2+15}
// C layout: VGPR r -> row r + 8*(lane>>4), col = lane&15
// ---------------------------------------------------------------------------
__device__ __forceinline__ v16h mk_frag(v8h lo, v8h hi) {
  v16h f;
#pragma unroll
  for (int i = 0; i < 8; ++i) { f[i] = lo[i]; f[i + 8] = hi[i]; }
  return f;
}

__device__ __forceinline__ v8f wmma16(v16h a, v16h b, v8f c) {
  return __builtin_amdgcn_wmma_f32_16x16x32_f16(false, a, false, b, (short)0, c,
                                                false, false);
}

// CDNA5 async DMA: global -> LDS, tracked by ASYNCcnt (CDNA5 ISA).
// Generic pointers to __shared__ carry the LDS offset in their low 32 bits.
__device__ __forceinline__ uint32_t lds_off(const void* p) {
  return (uint32_t)(uintptr_t)p;
}
__device__ __forceinline__ void async_b128(uint32_t dst_lds, const _Float16* g) {
  asm volatile("global_load_async_to_lds_b128 %0, %1, off"
               :: "v"(dst_lds), "v"(g) : "memory");
}
__device__ __forceinline__ void wait_async0() {
  asm volatile("s_wait_asynccnt 0" ::: "memory");
}

// ---------------------------------------------------------------------------
// f32 -> f16 conversion
// ---------------------------------------------------------------------------
__global__ void cvt_f32_f16(const float* __restrict__ src,
                            _Float16* __restrict__ dst, int n) {
  int i = blockIdx.x * blockDim.x + threadIdx.x;
  int stride = gridDim.x * blockDim.x;
  for (; i < n; i += stride) dst[i] = (_Float16)src[i];
}

// ---------------------------------------------------------------------------
// WMMA GEMM: C = (A[f16] @ B[f16] + bias) * scale
// Block tile 128x128, 8 waves as 4x2, wave tile 32x64, K-step 32.
// Double-buffered LDS; A tiles via global_load_async_to_lds_b128, B tiles
// software-pipelined through registers (needs K-transpose for frag reads).
// One workgroup barrier per K-step. M%128==0, N%128==0, K%32==0 assumed.
// ---------------------------------------------------------------------------
__global__ __launch_bounds__(256) void gemm_wmma(
    const _Float16* __restrict__ A, const _Float16* __restrict__ B,
    const float* __restrict__ bias, _Float16* __restrict__ outH,
    float* __restrict__ outF, int M, int N, int K, float scale) {
  __shared__ __align__(16) _Float16 sA[2][128 * 32];  // [m][k]
  __shared__ __align__(16) _Float16 sB[2][128 * 32];  // [n][k] (K-transposed)

  const int t = threadIdx.x;
  const int lane = t & 31;
  const int w = t >> 5;
  const int m0 = blockIdx.y * 128;
  const int n0 = blockIdx.x * 128;
  const int wr = (w >> 1) * 32;   // wave row offset in block tile
  const int wc = (w & 1) * 64;    // wave col offset in block tile
  const int lc = lane & 15;
  const int kb  = (lane >> 4) * 8;
  const int kb2 = (lane >> 4) * 16;

  const int ar = t >> 1, ac = (t & 1) * 16;  // A staging: 128 rows x 32 halves
  const int bk = t >> 4, bn = (t & 15) * 8;  // B staging: 2x16 rows x 128 halves

  v8f acc[2][4];
#pragma unroll
  for (int i = 0; i < 2; ++i)
#pragma unroll
    for (int j = 0; j < 4; ++j)
#pragma unroll
      for (int r = 0; r < 8; ++r) acc[i][j][r] = 0.f;

  const _Float16* Arow = A + (size_t)(m0 + ar) * K;

  // ---- prologue: stage K-step 0 into buffer 0
  async_b128(lds_off(&sA[0][ar * 32 + ac]), Arow + ac);
  async_b128(lds_off(&sA[0][ar * 32 + ac + 8]), Arow + ac + 8);
  {
    v8h b0 = *(const v8h*)(B + (size_t)bk * N + n0 + bn);
    v8h b1 = *(const v8h*)(B + (size_t)(bk + 16) * N + n0 + bn);
#pragma unroll
    for (int i = 0; i < 8; ++i) {
      sB[0][(bn + i) * 32 + bk] = b0[i];
      sB[0][(bn + i) * 32 + bk + 16] = b1[i];
    }
  }
  wait_async0();
  __syncthreads();

  int cur = 0;
  for (int k0 = 0; k0 < K; k0 += 32, cur ^= 1) {
    const int nxt = k0 + 32;
    v8h nb0, nb1;
    if (nxt < K) {
      // issue next tile's async DMA before compute (overlap with WMMA)
      async_b128(lds_off(&sA[cur ^ 1][ar * 32 + ac]), Arow + nxt + ac);
      async_b128(lds_off(&sA[cur ^ 1][ar * 32 + ac + 8]), Arow + nxt + ac + 8);
      nb0 = *(const v8h*)(B + (size_t)(nxt + bk) * N + n0 + bn);
      nb1 = *(const v8h*)(B + (size_t)(nxt + bk + 16) * N + n0 + bn);
    }
    // ---- compute: 2 A-frags x 4 B-frags = 8 WMMA
    const _Float16* sa = sA[cur];
    const _Float16* sb = sB[cur];
    v16h af[2];
#pragma unroll
    for (int mt = 0; mt < 2; ++mt) {
      const _Float16* pa = sa + (wr + mt * 16 + lc) * 32 + kb2;
      af[mt] = mk_frag(*(const v8h*)pa, *(const v8h*)(pa + 8));
    }
#pragma unroll
    for (int nt = 0; nt < 4; ++nt) {
      const _Float16* pbp = sb + (wc + nt * 16 + lc) * 32 + kb2;
      v16h bf = mk_frag(*(const v8h*)pbp, *(const v8h*)(pbp + 8));
#pragma unroll
      for (int mt = 0; mt < 2; ++mt)
        acc[mt][nt] = wmma16(af[mt], bf, acc[mt][nt]);
    }
    if (nxt < K) {
#pragma unroll
      for (int i = 0; i < 8; ++i) {
        sB[cur ^ 1][(bn + i) * 32 + bk] = nb0[i];
        sB[cur ^ 1][(bn + i) * 32 + bk + 16] = nb1[i];
      }
    }
    wait_async0();
    __syncthreads();
  }

  // ---- epilogue
#pragma unroll
  for (int mt = 0; mt < 2; ++mt) {
    const int row0 = m0 + wr + mt * 16 + ((lane >> 4) << 3);
#pragma unroll
    for (int nt = 0; nt < 4; ++nt) {
      int col = n0 + wc + nt * 16 + lc;
      float bv = bias[col];
#pragma unroll
      for (int r = 0; r < 8; ++r) {
        float v = (acc[mt][nt][r] + bv) * scale;
        size_t idx = (size_t)(row0 + r) * N + col;
        if (outH) outH[idx] = (_Float16)v;
        else      outF[idx] = v;
      }
    }
  }
}

// ---------------------------------------------------------------------------
// Banded local + global-key attention, flash-style online softmax.
// Block = (chunk c, head h); wave owns 16 query rows; 13 j-chunks of 32 cols
// (cols 0..31 = global keys 0..31, cols 32..415 = band keys).
// ---------------------------------------------------------------------------
__global__ __launch_bounds__(256) void band_attn(
    const _Float16* __restrict__ Qh, const _Float16* __restrict__ Kh,
    const _Float16* __restrict__ Vh, _Float16* __restrict__ AOh) {
  __shared__ __align__(16) _Float16 sProbs[8 * 16 * 32];  // per-wave 16x32
  __shared__ __align__(16) _Float16 sV[64 * 32];          // [d][j] transposed

  const int c = blockIdx.x;
  const int h = blockIdx.y;
  const int t = threadIdx.x;
  const int lane = t & 31;
  const int w = t >> 5;
  const int half = lane >> 4;
  const int lc = lane & 15;
  const int kb  = half * 8;
  const int kb2 = half * 16;

  // Q fragments (dh=64 -> two K-steps of 32); Q is pre-scaled by dh^-0.5
  const int qrowg = c * W_ + w * 16 + lc;
  v16h aq[2];
#pragma unroll
  for (int ks = 0; ks < 2; ++ks) {
    const _Float16* p = Qh + (size_t)qrowg * D_ + h * DH + ks * 32 + kb2;
    aq[ks] = mk_frag(*(const v8h*)p, *(const v8h*)(p + 8));
  }

  v8f outa[4];
  float m[8], l[8];
#pragma unroll
  for (int i = 0; i < 4; ++i)
#pragma unroll
    for (int j = 0; j < 8; ++j) outa[i][j] = 0.f;
#pragma unroll
  for (int r = 0; r < 8; ++r) { m[r] = -1e30f; l[r] = 0.f; }

  _Float16* pb = sProbs + w * (16 * 32);

  for (int chunk = 0; chunk < 13; ++chunk) {
    // ---- scores: Q @ K^T for this 32-col chunk (B frag direct from global K)
    v8f sc[2];
#pragma unroll
    for (int i = 0; i < 2; ++i)
#pragma unroll
      for (int j = 0; j < 8; ++j) sc[i][j] = 0.f;
#pragma unroll
    for (int tile = 0; tile < 2; ++tile) {
      int col = chunk * 32 + tile * 16 + lc;
      int j = (col < G_) ? col : (c * W_ - W_ + (col - G_));
      int jc = j < 0 ? 0 : (j > S_ - 1 ? S_ - 1 : j);
#pragma unroll
      for (int ks = 0; ks < 2; ++ks) {
        const _Float16* p = Kh + (size_t)jc * D_ + h * DH + ks * 32 + kb2;
        v16h bf = mk_frag(*(const v8h*)p, *(const v8h*)(p + 8));
        sc[tile] = wmma16(aq[ks], bf, sc[tile]);
      }
    }
    // ---- band mask (global cols always valid; reference masks band only)
#pragma unroll
    for (int tile = 0; tile < 2; ++tile) {
      int col = chunk * 32 + tile * 16 + lc;
#pragma unroll
      for (int r = 0; r < 8; ++r) {
        int qrow = w * 16 + half * 8 + r;  // 0..127 within chunk
        if (col >= G_) {
          int kcol = col - G_;
          int j = c * W_ - W_ + kcol;
          bool valid = (kcol >= qrow) && (kcol <= qrow + 2 * W_) &&
                       (j >= G_) && (j < S_);
          if (!valid) sc[tile][r] = -1e30f;
        }
      }
    }
    // ---- online softmax update (row reductions within 16-lane halves)
    float pr0[8], pr1[8];
#pragma unroll
    for (int r = 0; r < 8; ++r) {
      float cm = fmaxf(sc[0][r], sc[1][r]);
#pragma unroll
      for (int off = 1; off < 16; off <<= 1)
        cm = fmaxf(cm, __shfl_xor(cm, off, 32));
      float mnew = fmaxf(m[r], cm);
      float corr = __expf(m[r] - mnew);
      m[r] = mnew;
      l[r] *= corr;
#pragma unroll
      for (int dt = 0; dt < 4; ++dt) outa[dt][r] *= corr;
      pr0[r] = __expf(sc[0][r] - mnew);
      pr1[r] = __expf(sc[1][r] - mnew);
      float rs = pr0[r] + pr1[r];
#pragma unroll
      for (int off = 1; off < 16; off <<= 1)
        rs += __shfl_xor(rs, off, 32);
      l[r] += rs;
    }

    __syncthreads();  // previous chunk's sV reads complete before restage
    // ---- write probs (f16) to per-wave LDS; cooperatively stage V chunk
    {
#pragma unroll
      for (int r = 0; r < 8; ++r) {
        int row = half * 8 + r;
        pb[row * 32 + lc]      = (_Float16)pr0[r];
        pb[row * 32 + 16 + lc] = (_Float16)pr1[r];
      }
      int jr = t >> 3, d8 = (t & 7) * 8;
      int j = (chunk == 0) ? jr : (c * W_ - W_ + (chunk - 1) * 32 + jr);
      j = j < 0 ? 0 : (j > S_ - 1 ? S_ - 1 : j);
      v8h vv = *(const v8h*)(Vh + (size_t)j * D_ + h * DH + d8);
#pragma unroll
      for (int i = 0; i < 8; ++i) sV[(d8 + i) * 32 + jr] = vv[i];
    }
    __syncthreads();

    // ---- PV: out += probs(16x32) @ V(32x64), 4 WMMA over d tiles
    const _Float16* pa = pb + lc * 32 + kb2;
    v16h af = mk_frag(*(const v8h*)pa, *(const v8h*)(pa + 8));
#pragma unroll
    for (int dt = 0; dt < 4; ++dt) {
      const _Float16* pv = sV + (dt * 16 + lc) * 32 + kb2;
      v16h bf = mk_frag(*(const v8h*)pv, *(const v8h*)(pv + 8));
      outa[dt] = wmma16(af, bf, outa[dt]);
    }
  }

  // ---- normalize and store f16 attention output [S][D] head-major cols
#pragma unroll
  for (int dt = 0; dt < 4; ++dt) {
#pragma unroll
    for (int r = 0; r < 8; ++r) {
      int srow = c * W_ + w * 16 + half * 8 + r;
      int dcol = h * DH + dt * 16 + lc;
      AOh[(size_t)srow * D_ + dcol] = (_Float16)(outa[dt][r] / l[r]);
    }
  }
}

// ---------------------------------------------------------------------------
// Global-query attention (rows 0..G-1 attend to all S keys). Tiny workload.
// Block per (head, q-row): softmax over 4096 via workspace scores.
// ---------------------------------------------------------------------------
__global__ __launch_bounds__(256) void gq_scores(
    const _Float16* __restrict__ Qh, const _Float16* __restrict__ Kh,
    float* __restrict__ probs) {
  __shared__ float red[256];
  const int h = blockIdx.x >> 5;
  const int qi = blockIdx.x & 31;
  const int t = threadIdx.x;
  float* row = probs + (size_t)(h * G_ + qi) * S_;
  const _Float16* qp = Qh + (size_t)qi * D_ + h * DH;

  float lmax = -1e30f;
  for (int s = t; s < S_; s += 256) {
    const _Float16* kp = Kh + (size_t)s * D_ + h * DH;
    float acc = 0.f;
#pragma unroll
    for (int d = 0; d < DH; ++d) acc += (float)qp[d] * (float)kp[d];
    row[s] = acc;
    lmax = fmaxf(lmax, acc);
  }
  red[t] = lmax;
  __syncthreads();
  for (int sft = 128; sft > 0; sft >>= 1) {
    if (t < sft) red[t] = fmaxf(red[t], red[t + sft]);
    __syncthreads();
  }
  float mx = red[0];
  __syncthreads();
  float lsum = 0.f;
  for (int s = t; s < S_; s += 256) {
    float e = __expf(row[s] - mx);
    row[s] = e;
    lsum += e;
  }
  red[t] = lsum;
  __syncthreads();
  for (int sft = 128; sft > 0; sft >>= 1) {
    if (t < sft) red[t] += red[t + sft];
    __syncthreads();
  }
  float inv = 1.f / red[0];
  for (int s = t; s < S_; s += 256) row[s] *= inv;
}

__global__ __launch_bounds__(256) void gq_out(
    const float* __restrict__ probs, const _Float16* __restrict__ Vh,
    _Float16* __restrict__ AOh) {
  __shared__ float acc[256];
  const int h = blockIdx.x >> 5;
  const int qi = blockIdx.x & 31;
  const int t = threadIdx.x;
  const int d = t & 63;
  const int sg = t >> 6;
  const float* row = probs + (size_t)(h * G_ + qi) * S_;
  float part = 0.f;
  for (int s = sg * (S_ / 4); s < (sg + 1) * (S_ / 4); ++s)
    part += row[s] * (float)Vh[(size_t)s * D_ + h * DH + d];
  acc[t] = part;
  __syncthreads();
  if (t < 64) {
    float tot = acc[d] + acc[64 + d] + acc[128 + d] + acc[192 + d];
    AOh[(size_t)qi * D_ + h * DH + d] = (_Float16)tot;  // overwrite rows :G
  }
}

// ---------------------------------------------------------------------------
extern "C" void kernel_launch(void* const* d_in, const int* in_sizes, int n_in,
                              void* d_out, int out_size, void* d_ws,
                              size_t ws_size, hipStream_t stream) {
  (void)in_sizes; (void)n_in; (void)out_size; (void)ws_size;
  const float* hidden = (const float*)d_in[0];
  const float* Wq = (const float*)d_in[1];
  const float* bq = (const float*)d_in[2];
  const float* Wk = (const float*)d_in[3];
  const float* bk = (const float*)d_in[4];
  const float* Wv = (const float*)d_in[5];
  const float* bv = (const float*)d_in[6];
  const float* Wo = (const float*)d_in[7];
  const float* bo = (const float*)d_in[8];

  const size_t SD = (size_t)S_ * D_;   // 3,145,728
  const size_t DD = (size_t)D_ * D_;   //   589,824
  _Float16* hH  = (_Float16*)d_ws;
  _Float16* WqH = hH + SD;
  _Float16* WkH = WqH + DD;
  _Float16* WvH = WkH + DD;
  _Float16* WoH = WvH + DD;
  _Float16* Qh  = WoH + DD;
  _Float16* Kh  = Qh + SD;
  _Float16* Vh  = Kh + SD;
  _Float16* AOh = Vh + SD;
  float* gprobs = (float*)(AOh + SD);  // H*G*S f32

  cvt_f32_f16<<<2048, 256, 0, stream>>>(hidden, hH, (int)SD);
  cvt_f32_f16<<<1024, 256, 0, stream>>>(Wq, WqH, (int)DD);
  cvt_f32_f16<<<1024, 256, 0, stream>>>(Wk, WkH, (int)DD);
  cvt_f32_f16<<<1024, 256, 0, stream>>>(Wv, WvH, (int)DD);
  cvt_f32_f16<<<1024, 256, 0, stream>>>(Wo, WoH, (int)DD);

  dim3 gg(D_ / 128, S_ / 128);
  const float qscale = 0.125f;  // dh^-0.5, dh=64
  gemm_wmma<<<gg, 256, 0, stream>>>(hH, WqH, bq, Qh, nullptr, S_, D_, D_, qscale);
  gemm_wmma<<<gg, 256, 0, stream>>>(hH, WkH, bk, Kh, nullptr, S_, D_, D_, 1.0f);
  gemm_wmma<<<gg, 256, 0, stream>>>(hH, WvH, bv, Vh, nullptr, S_, D_, D_, 1.0f);

  band_attn<<<dim3(NC, H_), 256, 0, stream>>>(Qh, Kh, Vh, AOh);
  gq_scores<<<H_ * G_, 256, 0, stream>>>(Qh, Kh, gprobs);
  gq_out<<<H_ * G_, 256, 0, stream>>>(gprobs, Vh, AOh);

  gemm_wmma<<<gg, 256, 0, stream>>>(AOh, WoH, bo, nullptr, (float*)d_out,
                                    S_, D_, D_, 1.0f);
}
